// AdaptivePooling_45715631899658
// MI455X (gfx1250) — matrix operand order repaired and checked by reading, not compiled, not run on hardware
//
#include <hip/hip_runtime.h>
#include <math.h>

// Problem constants (fixed by setup_inputs)
#define Bn 16
#define Cn 128
#define Hn 192
#define Wn 192
#define HWn (Hn * Wn)        // 36864
#define HIDn 32
#define Kn 2

typedef __attribute__((ext_vector_type(16))) _Float16 v16h;
typedef __attribute__((ext_vector_type(8)))  float    v8f;
typedef __attribute__((ext_vector_type(4)))  int      v4i;

#if defined(__gfx1250__) && \
    __has_builtin(__builtin_amdgcn_global_load_async_to_lds_b128) && \
    __has_builtin(__builtin_amdgcn_s_wait_asynccnt)
#define USE_ASYNC_LDS 1
#else
#define USE_ASYNC_LDS 0
#endif

// ---------------------------------------------------------------------------
// Kernel 1: global average pool. One workgroup per (b,c) plane (2048 blocks).
// ---------------------------------------------------------------------------
__global__ __launch_bounds__(256) void dp_pool_kernel(
    const float* __restrict__ x, float* __restrict__ pooled)
{
    const int bc = blockIdx.x;
    const float4* p = (const float4*)(x + (size_t)bc * HWn);
    float s = 0.0f;
    for (int i = threadIdx.x; i < HWn / 4; i += 256) {
        float4 v = p[i];
        s += (v.x + v.y) + (v.z + v.w);
    }
#pragma unroll
    for (int off = 16; off > 0; off >>= 1)
        s += __shfl_down(s, off, 32);

    __shared__ float red[8];
    const int lane = threadIdx.x & 31;
    const int wid  = threadIdx.x >> 5;
    if (lane == 0) red[wid] = s;
    __syncthreads();
    if (threadIdx.x == 0) {
        float t = 0.0f;
#pragma unroll
        for (int i = 0; i < 8; ++i) t += red[i];
        pooled[bc] = t * (1.0f / (float)HWn);
    }
}

// ---------------------------------------------------------------------------
// Kernel 2: selector MLP. One wave (32 threads).
//   h    = relu(pooled[16,128] @ w1^T[128,32] + b1)   -- via v_wmma_f32_16x16x32_f16
//   attn = softmax(h @ w2^T + b2, axis=1)             -- VALU + shfl_xor
// Fragment layouts (wave32; A/B use contiguous k = 16*half + e, D per CDNA5 ISA §7.12.2):
//   A (16x32 f16): lane<16 -> M=lane, K(e) = base + e, base=0
//                  lane>=16 -> M=lane-16, base=16
//   B (32x16 f16): lane<16 -> N=lane,  K(e) = e ;  lane>=16 -> N=lane-16, K(e)=16+e
//   D (16x16 f32): vgpr r, lane l -> M = r + 8*(l>=16), N = l%16
// ---------------------------------------------------------------------------
__global__ __launch_bounds__(32) void dp_selector_kernel(
    const float* __restrict__ pooled,
    const float* __restrict__ w1p, const float* __restrict__ b1p,
    const float* __restrict__ w2p, const float* __restrict__ b2p,
    float* __restrict__ attn_out)
{
    const int lane = threadIdx.x;
    const int hv   = lane >> 4;   // 0 or 1 (lane half)
    const int lm   = lane & 15;

    v8f acc0 = {};
    v8f acc1 = {};

#pragma unroll
    for (int kk = 0; kk < 4; ++kk) {           // K = 128 in chunks of 32
        const int kbaseA = kk * 32 + hv * 16;
        v16h a;
#pragma unroll
        for (int e = 0; e < 16; ++e) {
            const int k = kbaseA + e;
            a[e] = (_Float16)pooled[lm * Cn + k];
        }
        const int kbaseB = kk * 32 + hv * 16;
        v16h bf0, bf1;
#pragma unroll
        for (int e = 0; e < 16; ++e) {
            const int k = kbaseB + e;
            bf0[e] = (_Float16)w1p[(lm)      * Cn + k];  // N tile 0: hid = lm
            bf1[e] = (_Float16)w1p[(16 + lm) * Cn + k];  // N tile 1: hid = 16+lm
        }
        acc0 = __builtin_amdgcn_wmma_f32_16x16x32_f16(
            false, a, false, bf0, (short)0, acc0, false, false);
        acc1 = __builtin_amdgcn_wmma_f32_16x16x32_f16(
            false, a, false, bf1, (short)0, acc1, false, false);
    }

    // bias + relu, scatter D-layout into LDS as h[batch][hid]
    __shared__ float hbuf[Bn][HIDn];
#pragma unroll
    for (int r = 0; r < 8; ++r) {
        const int batch = r + hv * 8;
        hbuf[batch][lm]      = fmaxf(acc0[r] + b1p[lm],      0.0f);
        hbuf[batch][16 + lm] = fmaxf(acc1[r] + b1p[16 + lm], 0.0f);
    }
    __syncthreads();

    // logits: lane -> (b = lm, k = hv); then 2-way softmax across lane^16
    float z = b2p[hv];
#pragma unroll
    for (int j = 0; j < HIDn; ++j)
        z += hbuf[lm][j] * w2p[hv * HIDn + j];

    const float zo = __shfl_xor(z, 16, 32);
    const float m  = fmaxf(z, zo);
    const float e0 = __expf(z - m);
    const float e1 = __expf(zo - m);
    attn_out[lm * Kn + hv] = e0 / (e0 + e1);
}

// ---------------------------------------------------------------------------
// Kernel 3: blended depthwise 3x3 conv + bias. One workgroup per
// (b, c, 16-row tile): stage 18 rows x 192 cols (13.8 KB) into LDS via
// async global->LDS (ASYNCcnt), then each thread produces 3 float4 outputs.
// ---------------------------------------------------------------------------
__global__ __launch_bounds__(256) void dp_conv_kernel(
    const float* __restrict__ x, const float* __restrict__ attn,
    const float* __restrict__ bank, const float* __restrict__ bias,
    float* __restrict__ out)
{
    const int tpp    = Hn / 16;                 // 12 tiles per plane
    const int blk    = blockIdx.x;
    const int bc     = blk / tpp;
    const int tile_i = blk - bc * tpp;
    const int b      = bc >> 7;                 // / Cn
    const int c      = bc & (Cn - 1);
    const int y0     = tile_i * 16;
    const int tid    = threadIdx.x;

    // blended per-(b,c) 3x3 kernel (uniform across the block)
    const float a0 = attn[b * Kn + 0];
    const float a1 = attn[b * Kn + 1];
    float w9[9];
#pragma unroll
    for (int i = 0; i < 9; ++i)
        w9[i] = a0 * bank[c * 9 + i] + a1 * bank[Cn * 9 + c * 9 + i];
    const float bias_c = bias[c];

    // stage rows [y0-1, y0+16] of this plane into LDS (contiguous slab)
    __shared__ __align__(16) float tile[18 * Wn];
    const float* plane   = x + (size_t)bc * HWn;
    const int gy_start   = (y0 == 0) ? 0 : (y0 - 1);
    const int gy_end     = (y0 + 17 > Hn) ? Hn : (y0 + 17);
    const int nrows      = gy_end - gy_start;
    const int lr0        = gy_start - (y0 - 1);        // 0 or 1
    const float* gsrc    = plane + (size_t)gy_start * Wn;
    float* lbase         = &tile[lr0 * Wn];
    const int nvec       = nrows * (Wn / 4);

#if USE_ASYNC_LDS
    for (int i = tid; i < nvec; i += 256) {
        __builtin_amdgcn_global_load_async_to_lds_b128(
            (__attribute__((address_space(1))) v4i*)(gsrc + 4 * i),
            (__attribute__((address_space(3))) v4i*)(lbase + 4 * i),
            0, 0);
    }
    __builtin_amdgcn_s_wait_asynccnt(0);
#else
    {
        const float4* g4 = (const float4*)gsrc;
        float4* l4 = (float4*)lbase;
        for (int i = tid; i < nvec; i += 256) l4[i] = g4[i];
    }
#endif
    __syncthreads();

    // each thread computes 3 float4 output vectors
#pragma unroll
    for (int it = 0; it < 3; ++it) {
        const int v  = tid + it * 256;
        const int ry = v / (Wn / 4);
        const int xv = v - ry * (Wn / 4);
        const int x0 = xv * 4;
        const int y  = y0 + ry;

        float acc0 = bias_c, acc1 = bias_c, acc2 = bias_c, acc3 = bias_c;
#pragma unroll
        for (int dy = -1; dy <= 1; ++dy) {
            const int gy = y + dy;
            if (gy < 0 || gy >= Hn) continue;
            const float* row = &tile[(gy - (y0 - 1)) * Wn];
            const float lv = (x0 > 0)       ? row[x0 - 1] : 0.0f;
            const float c0 = row[x0];
            const float c1 = row[x0 + 1];
            const float c2 = row[x0 + 2];
            const float c3 = row[x0 + 3];
            const float rv = (x0 + 4 < Wn)  ? row[x0 + 4] : 0.0f;
            const float k0 = w9[(dy + 1) * 3 + 0];
            const float k1 = w9[(dy + 1) * 3 + 1];
            const float k2 = w9[(dy + 1) * 3 + 2];
            acc0 += k0 * lv + k1 * c0 + k2 * c1;
            acc1 += k0 * c0 + k1 * c1 + k2 * c2;
            acc2 += k0 * c1 + k1 * c2 + k2 * c3;
            acc3 += k0 * c2 + k1 * c3 + k2 * rv;
        }
        float4 o = make_float4(acc0, acc1, acc2, acc3);
        *(float4*)(out + (size_t)bc * HWn + (size_t)y * Wn + x0) = o;
    }
}

// ---------------------------------------------------------------------------
extern "C" void kernel_launch(void* const* d_in, const int* in_sizes, int n_in,
                              void* d_out, int out_size, void* d_ws, size_t ws_size,
                              hipStream_t stream)
{
    const float* x    = (const float*)d_in[0];
    const float* w1   = (const float*)d_in[1];
    const float* b1   = (const float*)d_in[2];
    const float* w2   = (const float*)d_in[3];
    const float* b2   = (const float*)d_in[4];
    const float* bank = (const float*)d_in[5];
    const float* bias = (const float*)d_in[6];
    float* out = (float*)d_out;

    float* ws     = (float*)d_ws;
    float* pooled = ws;              // [16*128]
    float* attn   = ws + Bn * Cn;    // [16*2]

    dp_pool_kernel<<<Bn * Cn, 256, 0, stream>>>(x, pooled);
    dp_selector_kernel<<<1, 32, 0, stream>>>(pooled, w1, b1, w2, b2, attn);
    dp_conv_kernel<<<Bn * Cn * (Hn / 16), 256, 0, stream>>>(x, attn, bank, bias, out);
}
